// MultiheadDistanceSelfAttention_1881195676264
// MI455X (gfx1250) — matrix operand order repaired and checked by reading, not compiled, not run on hardware
//
#include <hip/hip_runtime.h>
#include <hip/hip_bf16.h>
#include <math.h>

typedef __attribute__((ext_vector_type(16))) _Float16 v16h;
typedef __attribute__((ext_vector_type(2)))  _Float16 h2;
typedef __attribute__((ext_vector_type(8)))  float    v8f;

#define B_   2
#define N_   192
#define D_   256
#define DH_  1024           // 4*D hidden width
#define SCALE_ 0.1767766952966369f   // 32^-0.5

// workspace layout (bytes)
#define WS_W2_OFF   0                            // 262144 halves = 512 KB (f16 B-fragments)
#define WS_QKV_OFF  524288                       // 384*768 f32   = 1.125 MB
#define WS_OPRE_OFF (524288 + 1179648)           // 384*256 f32   = 384 KB
#define WS_W1P_OFF  (524288 + 1179648 + 393216)  // 1024 halves   = 2 KB
#define WS_B1P_OFF  (WS_W1P_OFF + 2048)          // 1024 halves   = 2 KB

// ---------------------------------------------------------------------------
// Kernel 0a: repack w_mlp2 (1024x256 f32, row-major) into f16 WMMA B-fragments.
// Fragment index f = ((ct*32 + kk)*32 + lane)*16 + e
//   column n = ct*16 + (lane&15);  K = kk*32 + (lane>>4)*16 + e
// (contiguous K order, same as the A fragments: lanes 0-15 hold K 0..15, lanes 16-31 K 16..31)
// ---------------------------------------------------------------------------
__global__ __launch_bounds__(256) void k_w2_to_f16(const float* __restrict__ w2,
                                                   _Float16* __restrict__ w2h) {
    int f    = blockIdx.x * 256 + threadIdx.x;   // [0, 262144)
    int e    = f & 15;
    int lane = (f >> 4) & 31;
    int kk   = (f >> 9) & 31;
    int ct   = f >> 14;
    int K    = kk * 32 + (lane >> 4) * 16 + e;
    int n    = ct * 16 + (lane & 15);
    w2h[f] = (_Float16)w2[K * 256 + n];
}

// ---------------------------------------------------------------------------
// Kernel 0b: prepack w_mlp1 / b_mlp1 into A-fragment element order as f16, so
// the hidden fill can run on v_pk_fma_f16 with contiguous 32 B reads.
// idx = kk*32 + hi*16 + e  ->  source K = kk*32 + 16*hi + e
// ---------------------------------------------------------------------------
__global__ __launch_bounds__(256) void k_w1_pack(const float* __restrict__ w1,
                                                 const float* __restrict__ b1,
                                                 _Float16* __restrict__ w1p,
                                                 _Float16* __restrict__ b1p) {
    int idx = blockIdx.x * 256 + threadIdx.x;    // [0, 1024)
    int e   = idx & 15;
    int hi  = (idx >> 4) & 1;
    int kk  = idx >> 5;
    int K   = kk * 32 + (hi << 4) + e;
    w1p[idx] = (_Float16)w1[K];
    b1p[idx] = (_Float16)b1[K];
}

// ---------------------------------------------------------------------------
// Kernel 1: qkv = x @ w_qkv + b_qkv        (384 x 256) @ (256 x 768)
// ---------------------------------------------------------------------------
__global__ __launch_bounds__(256) void k_qkv(const float* __restrict__ x,
                                             const float* __restrict__ w,
                                             const float* __restrict__ bq,
                                             float* __restrict__ qkv) {
    __shared__ float xr[256];
    const int row = blockIdx.x;        // b*N + n
    const int tid = threadIdx.x;
    xr[tid] = x[row * 256 + tid];
    __syncthreads();
    float a0 = bq[tid], a1 = bq[tid + 256], a2 = bq[tid + 512];
    for (int k = 0; k < 256; ++k) {
        const float xv = xr[k];
        const float* wr = w + k * 768;
        a0 = fmaf(xv, wr[tid],       a0);
        a1 = fmaf(xv, wr[tid + 256], a1);
        a2 = fmaf(xv, wr[tid + 512], a2);
    }
    qkv[row * 768 + tid]       = a0;
    qkv[row * 768 + tid + 256] = a1;
    qkv[row * 768 + tid + 512] = a2;
}

// ---------------------------------------------------------------------------
// Kernel 2: fused demb-GEMM (WMMA f16->f32) + score + online softmax + V acc.
// One workgroup per (b, i); 8 waves; wave w owns channel tiles {2w, 2w+1}.
// ---------------------------------------------------------------------------
__global__ __launch_bounds__(256) void k_attn(const float* __restrict__ pos,
                                              const unsigned char* __restrict__ mask, // jnp bool = 1 byte
                                              const h2* __restrict__ w1p,
                                              const h2* __restrict__ b1p,
                                              const float* __restrict__ b2,
                                              const _Float16* __restrict__ w2h,
                                              const float* __restrict__ qkv,
                                              float* __restrict__ opre) {
    __shared__ __align__(32) _Float16 ldsA[16 * DH_];   // 32 KB A-fragments
    __shared__ h2    s_w1p[512];                        // 2 KB, fragment-ordered
    __shared__ h2    s_b1p[512];                        // 2 KB
    __shared__ float s_dist[16];
    __shared__ float s_decay[16];
    __shared__ int   s_maskj[16];

    const int i    = blockIdx.x;
    const int b    = blockIdx.y;
    const int tid  = threadIdx.x;
    const int lane = tid & 31;
    const int wave = tid >> 5;
    const int col  = lane & 15;
    const int hi   = lane >> 4;

    for (int t = tid; t < 512; t += 256) { s_w1p[t] = w1p[t]; s_b1p[t] = b1p[t]; }

    const int   rowI = b * N_ + i;
    const float pix = pos[rowI * 3 + 0];
    const float piy = pos[rowI * 3 + 1];
    const float piz = pos[rowI * 3 + 2];
    const int   mi  = mask[rowI];

    // this lane's two channels
    const int   d0  = wave * 32 + col;      // = (2*wave)*16 + col
    const int   d1  = d0 + 16;
    const float k0  = qkv[rowI * 768 + 256 + d0];
    const float k1  = qkv[rowI * 768 + 256 + d1];
    const float bb0 = b2[d0], bb1 = b2[d1];

    // online-softmax state per owned channel
    float m0 = -INFINITY, m1 = -INFINITY;
    float su0 = 0.f, su1 = 0.f;
    float o0 = 0.f, o1 = 0.f;

    const v16h* Af = (const v16h*)ldsA;
    const v16h* Bf = (const v16h*)w2h;
    const int ct0 = wave * 2, ct1 = ct0 + 1;
    const h2 zero2 = {(_Float16)0.f, (_Float16)0.f};

    for (int jt = 0; jt < N_ / 16; ++jt) {
        const int j0 = jt * 16;
        __syncthreads();                       // prior epilogue done with s_* / ldsA
        if (tid < 16) {
            const int   rj = b * N_ + j0 + tid;
            const float dx = pix - pos[rj * 3 + 0];
            const float dy = piy - pos[rj * 3 + 1];
            const float dz = piz - pos[rj * 3 + 2];
            const float d2 = dx * dx + dy * dy + dz * dz;
            s_dist[tid]  = sqrtf(d2);
            s_decay[tid] = __expf(-d2);        // DIST_DECAY = 1.0
            s_maskj[tid] = mask[rj];
        }
        __syncthreads();

        // relu-hidden tile (16 j x 1024 h) straight into A-fragment layout,
        // computed in packed f16 (v_pk_fma_f16 / v_pk_max): slot s = (kk,lane),
        // each thread fills 4 slots x 16 halves, one b256 LDS store per slot.
        #pragma unroll
        for (int ss = 0; ss < 4; ++ss) {
            const int s    = tid * 4 + ss;
            const int kk   = s >> 5;
            const int ln   = s & 31;
            const _Float16 th = (_Float16)s_dist[ln & 15];
            const h2  t2   = {th, th};
            const int base = kk * 16 + ((ln >> 4) << 3);   // h2 units
            v16h val;
            h2* vp = (h2*)&val;
            #pragma unroll
            for (int e2 = 0; e2 < 8; ++e2) {
                h2 h = t2 * s_w1p[base + e2] + s_b1p[base + e2];
                vp[e2] = __builtin_elementwise_max(h, zero2);
            }
            *(v16h*)&ldsA[s * 16] = val;
        }
        __syncthreads();

        // demb tile: 32 K-steps x 2 channel tiles of v_wmma_f32_16x16x32_f16
        v8f c0 = {}; v8f c1 = {};
        #pragma unroll 4
        for (int kk = 0; kk < 32; ++kk) {
            const v16h a   = Af[kk * 32 + lane];
            const v16h bf0 = Bf[(ct0 * 32 + kk) * 32 + lane];
            const v16h bf1 = Bf[(ct1 * 32 + kk) * 32 + lane];
            c0 = __builtin_amdgcn_wmma_f32_16x16x32_f16(false, a, false, bf0,
                                                        (short)0, c0, false, false);
            c1 = __builtin_amdgcn_wmma_f32_16x16x32_f16(false, a, false, bf1,
                                                        (short)0, c1, false, false);
        }

        // fused epilogue: C row M = r + 8*hi (= j), col N = lane&15 (= d)
        #pragma unroll
        for (int r = 0; r < 8; ++r) {
            const int   jr   = r + (hi << 3);
            const int   j    = j0 + jr;
            const float dec  = s_decay[jr];
            const int   dead = mi & s_maskj[jr];
            const float* qj  = qkv + (b * N_ + j) * 768;
            {   // channel d0
                const float q  = qj[d0];
                const float vv = qj[512 + d0];
                const float sc = dec * (k0 + bb0 + c0[r]) * q * SCALE_;
                if (!dead) {
                    const float mn = fmaxf(m0, sc);
                    const float cr = __expf(m0 - mn);
                    const float p  = __expf(sc - mn);
                    su0 = su0 * cr + p;
                    o0  = o0  * cr + p * vv;
                    m0  = mn;
                }
            }
            {   // channel d1
                const float q  = qj[d1];
                const float vv = qj[512 + d1];
                const float sc = dec * (k1 + bb1 + c1[r]) * q * SCALE_;
                if (!dead) {
                    const float mn = fmaxf(m1, sc);
                    const float cr = __expf(m1 - mn);
                    const float p  = __expf(sc - mn);
                    su1 = su1 * cr + p;
                    o1  = o1  * cr + p * vv;
                    m1  = mn;
                }
            }
        }
    }

    // merge the two row-halves (lane L holds j rows 0-7, lane L^16 rows 8-15)
    {
        const float m2 = __shfl_xor(m0, 16, 32);
        const float s2 = __shfl_xor(su0, 16, 32);
        const float q2 = __shfl_xor(o0, 16, 32);
        const float Mx = fmaxf(m0, m2);
        const float e1 = __expf(m0 - Mx), e2 = __expf(m2 - Mx);
        const float S  = su0 * e1 + s2 * e2;
        const float O  = o0  * e1 + q2 * e2;
        if (hi == 0) opre[rowI * 256 + d0] = O / S;
    }
    {
        const float m2 = __shfl_xor(m1, 16, 32);
        const float s2 = __shfl_xor(su1, 16, 32);
        const float q2 = __shfl_xor(o1, 16, 32);
        const float Mx = fmaxf(m1, m2);
        const float e1 = __expf(m1 - Mx), e2 = __expf(m2 - Mx);
        const float S  = su1 * e1 + s2 * e2;
        const float O  = o1  * e1 + q2 * e2;
        if (hi == 0) opre[rowI * 256 + d1] = O / S;
    }
}

// ---------------------------------------------------------------------------
// Kernel 3: out = opre @ w_out + b_out     (384 x 256) @ (256 x 256)
// ---------------------------------------------------------------------------
__global__ __launch_bounds__(256) void k_out(const float* __restrict__ opre,
                                             const float* __restrict__ wout,
                                             const float* __restrict__ bout,
                                             float* __restrict__ out) {
    __shared__ float r[256];
    const int row = blockIdx.x;
    const int tid = threadIdx.x;
    r[tid] = opre[row * 256 + tid];
    __syncthreads();
    float acc = bout[tid];
    for (int k = 0; k < 256; ++k)
        acc = fmaf(r[k], wout[k * 256 + tid], acc);
    out[row * 256 + tid] = acc;
}

extern "C" void kernel_launch(void* const* d_in, const int* in_sizes, int n_in,
                              void* d_out, int out_size, void* d_ws, size_t ws_size,
                              hipStream_t stream) {
    const float*         x      = (const float*)d_in[0];
    const float*         pos    = (const float*)d_in[1];
    const unsigned char* mask   = (const unsigned char*)d_in[2];   // jnp bool = 1 byte
    const float*         w_qkv  = (const float*)d_in[3];
    const float*         b_qkv  = (const float*)d_in[4];
    const float*         w_mlp1 = (const float*)d_in[5];
    const float*         b_mlp1 = (const float*)d_in[6];
    const float*         w_mlp2 = (const float*)d_in[7];
    const float*         b_mlp2 = (const float*)d_in[8];
    const float*         w_out  = (const float*)d_in[9];
    const float*         b_out  = (const float*)d_in[10];

    char*      ws   = (char*)d_ws;            // needs ~2.1 MB
    _Float16*  w2h  = (_Float16*)(ws + WS_W2_OFF);
    float*     qkv  = (float*)(ws + WS_QKV_OFF);
    float*     opre = (float*)(ws + WS_OPRE_OFF);
    _Float16*  w1p  = (_Float16*)(ws + WS_W1P_OFF);
    _Float16*  b1p  = (_Float16*)(ws + WS_B1P_OFF);

    k_w2_to_f16<<<1024, 256, 0, stream>>>(w_mlp2, w2h);
    k_w1_pack<<<4, 256, 0, stream>>>(w_mlp1, b_mlp1, w1p, b1p);
    k_qkv<<<B_ * N_, 256, 0, stream>>>(x, w_qkv, b_qkv, qkv);
    k_attn<<<dim3(N_, B_), 256, 0, stream>>>(pos, mask, (const h2*)w1p,
                                             (const h2*)b1p, b_mlp2,
                                             w2h, qkv, opre);
    k_out<<<B_ * N_, 256, 0, stream>>>(opre, w_out, b_out, (float*)d_out);
}
